// NeuralSearchModule_30631706755214
// MI455X (gfx1250) — matrix operand order repaired and checked by reading, not compiled, not run on hardware
//
#include <hip/hip_runtime.h>

typedef __attribute__((ext_vector_type(16))) _Float16 v16h;
typedef __attribute__((ext_vector_type(8)))  float    v8f;

#define BATCH 8
#define NN    256
#define DD    128
#define TWO_D 256

// Workspace layout:
//   [0, 2MB)        PiB = x@W1a + roll(x)@W1b + b1   [B][N][2D] fp32
//   [2MB, 4MB)      Pj  = x@W1c + roll(x)@W1d        [B][N][2D] fp32
//   [4MB, 4MB+64KB) W2h : W2 as f16, pre-swizzled into WMMA B-fragment order

// ---------------------------------------------------------------------------
// Kernel 1: per-row projections (tiny GEMMs, ~0.5 GFLOP total).
// One block per (b,n) row; thread t = output channel c in [0,256).
// ---------------------------------------------------------------------------
__global__ void prep_proj_kernel(const float* __restrict__ x,
                                 const float* __restrict__ W1,
                                 const float* __restrict__ b1,
                                 float* __restrict__ PiB,
                                 float* __restrict__ Pj) {
  __shared__ float xr[DD];
  __shared__ float xnr[DD];
  const int bn = blockIdx.x;
  const int b = bn >> 8, n = bn & 255;
  const int t = threadIdx.x;
  if (t < DD) {
    xr[t] = x[bn * DD + t];
  } else {
    const int d = t - DD;
    xnr[d] = x[(b * NN + ((n + 1) & 255)) * DD + d];
  }
  __syncthreads();
  float ai = b1[t], aj = 0.0f;
#pragma unroll 4
  for (int d = 0; d < DD; ++d) {
    const float xv = xr[d], xnv = xnr[d];
    ai = fmaf(xv,  W1[d * TWO_D + t],            ai);
    ai = fmaf(xnv, W1[(DD + d) * TWO_D + t],     ai);
    aj = fmaf(xv,  W1[(2 * DD + d) * TWO_D + t], aj);
    aj = fmaf(xnv, W1[(3 * DD + d) * TWO_D + t], aj);
  }
  PiB[bn * TWO_D + t] = ai;
  Pj[bn * TWO_D + t]  = aj;
}

// ---------------------------------------------------------------------------
// Kernel 2: convert W2 [2D,D] fp32 -> f16, pre-swizzled into the exact per-lane
// WMMA B-fragment layout (fragment (ks,nt): lane L holds N = nt*16 + L%16,
// halves idx 0..15 hold K = ks*32 + (idx&7) + 8*(L>=16) + 16*(idx>=8)).  Also writes the
// trivial improved_tour output.
// ---------------------------------------------------------------------------
__global__ void prep_pack_kernel(const float* __restrict__ W2,
                                 _Float16* __restrict__ W2h,
                                 int* __restrict__ tour) {
  const int o = blockIdx.x * blockDim.x + threadIdx.x;   // 0..32767
  const int idx  = o & 15;
  const int lane = (o >> 4) & 31;
  const int frag = o >> 9;                // ks*8 + nt
  const int nt = frag & 7, ks = frag >> 3;
  const int k = ks * 32 + (idx & 7) + ((lane >= 16) ? 8 : 0) + ((idx >= 8) ? 16 : 0);
  const int n = nt * 16 + (lane & 15);
  W2h[o] = (_Float16)W2[k * DD + n];
  if (o < BATCH * NN) tour[o] = o & 255;
}

// ---------------------------------------------------------------------------
// Kernel 3: fused pairwise scorer.  One block = one (b, 16x16 (i,j) tile).
// 8 waves; wave w owns M-tiles {2w, 2w+1} (i = i0+mt fixed per tile,
// j = j0 + lane%16) and all 8 N-tiles of the D=128 h2 channels.
// h1 is built in registers from LDS-staged PiB/Pj (forced ds_load_b128 via
// 8-wide vector reads), fed to v_wmma_f32_16x16x32_f16 against LDS-staged
// pre-swizzled W2 fragments.  Epilogue does relu(+b2) . W3, cross-lane
// reduce, tanh, mask, store.
//
// Dynamic LDS partition (98304 bytes total):
//   [0,      65536)  W2_s  : 32768 f16 fragments
//   [65536,  81920)  PiB_s : 16 rows x 256 fp32
//   [81920,  98304)  Pj_s  : 16 rows x 256 fp32
// ---------------------------------------------------------------------------
__global__ __launch_bounds__(256)
void pair_score_kernel(const float* __restrict__ PiB_g,
                       const float* __restrict__ Pj_g,
                       const _Float16* __restrict__ W2h,
                       const float* __restrict__ b2,
                       const float* __restrict__ W3,
                       const float* __restrict__ b3,
                       float* __restrict__ outImp) {
  extern __shared__ char smem[];
  _Float16* W2_s  = (_Float16*)smem;             // 64 KB
  float*    PiB_s = (float*)(smem + 65536);      // 16 KB
  float*    Pj_s  = (float*)(smem + 81920);      // 16 KB

  const int jT = blockIdx.x, iT = blockIdx.y, b = blockIdx.z;
  const int i0 = iT * 16, j0 = jT * 16;
  const int t = threadIdx.x;
  const int lane = t & 31, wave = t >> 5;
  const int l16 = lane & 15, hiHalf = lane >> 4;

  // Stage the first 32 KB of the pre-swizzled W2 into LDS (straight vector copy).
  {
    const uint4* src = (const uint4*)W2h + t * 8;
    uint4* dst = (uint4*)W2_s + t * 8;
#pragma unroll
    for (int q = 0; q < 8; ++q) dst[q] = src[q];
  }
  // Stage 16 PiB rows (i-side) and 16 Pj rows (j-side) into LDS, coalesced.
  {
    const int row = t >> 4, seg = (t & 15) * 16;
    const float4* sPi = (const float4*)&PiB_g[((b * NN) + i0 + row) * TWO_D + seg];
    const float4* sPj = (const float4*)&Pj_g[((b * NN) + j0 + row) * TWO_D + seg];
    float4* dPi = (float4*)&PiB_s[row * TWO_D + seg];
    float4* dPj = (float4*)&Pj_s[row * TWO_D + seg];
#pragma unroll
    for (int q = 0; q < 4; ++q) { dPi[q] = sPi[q]; dPj[q] = sPj[q]; }
  }

  // Per-lane epilogue constants: this lane always sits at column n%16 == l16.
  float b2v[8], w3v[8];
#pragma unroll
  for (int nt = 0; nt < 8; ++nt) {
    b2v[nt] = b2[nt * 16 + l16];
    w3v[nt] = W3[nt * 16 + l16];
  }
  const float b3v = b3[0];
  __syncthreads();

  const float* pjRow = &Pj_s[l16 * TWO_D];
#pragma unroll
  for (int mt2 = 0; mt2 < 2; ++mt2) {
    const int mt = wave * 2 + mt2;
    const float* piRow = &PiB_s[mt * TWO_D];

    v8f acc[8];
#pragma unroll
    for (int nt = 0; nt < 8; ++nt)
#pragma unroll
      for (int e = 0; e < 8; ++e) acc[nt][e] = 0.0f;

#pragma unroll
    for (int ks = 0; ks < 8; ++ks) {
      // A-fragment layout (16-bit A 16x32): lane half selects K sub-range.
      // Each 8-float run is 32B-aligned -> force 2x ds_load_b128 per run.
      const int r0 = ks * 32 + (hiHalf ? 8 : 0);
      const v8f pi_lo = *(const v8f*)(piRow + r0);
      const v8f pi_hi = *(const v8f*)(piRow + r0 + 16);
      const v8f pj_lo = *(const v8f*)(pjRow + r0);
      const v8f pj_hi = *(const v8f*)(pjRow + r0 + 16);
      const v8f lo = pi_lo + pj_lo;   // v_pk_add_f32
      const v8f hi = pi_hi + pj_hi;
      v16h a;
#pragma unroll
      for (int h = 0; h < 8; ++h) {
        a[h]     = (_Float16)fmaxf(lo[h], 0.0f);   // relu(Pi + Pj + b1)
        a[h + 8] = (_Float16)fmaxf(hi[h], 0.0f);
      }
      const v16h* bBase = (const v16h*)W2_s + (ks * 8) * 32 + lane;
#pragma unroll
      for (int nt = 0; nt < 8; ++nt) {
        const v16h bf = bBase[nt * 32];   // stride: 32 lanes * 16 halves
        acc[nt] = __builtin_amdgcn_wmma_f32_16x16x32_f16(
            false, a, false, bf, (short)0, acc[nt], false, false);
      }
    }

    // Epilogue: relu(h2 + b2) . W3, reduce 16 lanes (one D-row per lane half).
    float partial[8];
#pragma unroll
    for (int r = 0; r < 8; ++r) partial[r] = 0.0f;
#pragma unroll
    for (int nt = 0; nt < 8; ++nt)
#pragma unroll
      for (int r = 0; r < 8; ++r)
        partial[r] += fmaxf(acc[nt][r] + b2v[nt], 0.0f) * w3v[nt];
#pragma unroll
    for (int r = 0; r < 8; ++r) {
      partial[r] += __shfl_xor(partial[r], 1, 32);
      partial[r] += __shfl_xor(partial[r], 2, 32);
      partial[r] += __shfl_xor(partial[r], 4, 32);
      partial[r] += __shfl_xor(partial[r], 8, 32);
    }
    float sval = partial[0];
#pragma unroll
    for (int r = 1; r < 8; ++r) sval = (l16 == r) ? partial[r] : sval;
    if (l16 < 8) {
      const int m_row = l16 + hiHalf * 8;          // D row = local j
      const int i = i0 + mt, j = j0 + m_row;
      const float val = tanhf(sval + b3v);
      const bool valid = (j >= i + 2) && ((j - i) != (NN - 1));
      outImp[((b * NN) + i) * NN + j] = valid ? val : 0.0f;
    }
  }
}

// ---------------------------------------------------------------------------
extern "C" void kernel_launch(void* const* d_in, const int* in_sizes, int n_in,
                              void* d_out, int out_size, void* d_ws, size_t ws_size,
                              hipStream_t stream) {
  (void)in_sizes; (void)n_in; (void)out_size; (void)ws_size;
  const float* x  = (const float*)d_in[0];
  const float* W1 = (const float*)d_in[1];
  const float* b1 = (const float*)d_in[2];
  const float* W2 = (const float*)d_in[3];
  const float* b2 = (const float*)d_in[4];
  const float* W3 = (const float*)d_in[5];
  const float* b3 = (const float*)d_in[6];

  int*   tour = (int*)d_out;                       // [B,N] int32
  float* imp  = (float*)d_out + BATCH * NN;        // [B,N,N] fp32

  char* ws = (char*)d_ws;
  float*    PiB = (float*)ws;                                  // 2 MB
  float*    Pj  = (float*)(ws + (size_t)2 * 1024 * 1024);      // 2 MB
  _Float16* W2h = (_Float16*)(ws + (size_t)4 * 1024 * 1024);   // 64 KB

  prep_proj_kernel<<<BATCH * NN, 256, 0, stream>>>(x, W1, b1, PiB, Pj);
  prep_pack_kernel<<<(TWO_D * DD) / 256, 256, 0, stream>>>(W2, W2h, tour);

  dim3 grid(16, 16, BATCH);   // (jTile, iTile, batch)
  pair_score_kernel<<<grid, 256, 98304, stream>>>(PiB, Pj, W2h, b2, W3, b3, imp);
}
